// MoEFFN_42949673735
// MI455X (gfx1250) — matrix operand order repaired and checked by reading, not compiled, not run on hardware
//
#include <hip/hip_runtime.h>

#define NUM_TOK   4096
#define DMODEL    1024
#define DINNER    2048
#define NEXP      8
#define SLOTS_PAD (2*NUM_TOK + NEXP*32)   /* 8448: per-expert buckets padded to 32 */
#define MTILES    (SLOTS_PAD/32)          /* 264: 32-slot M tiles */

typedef __bf16 bf16;
typedef __attribute__((ext_vector_type(16))) __bf16 v16bf;
typedef __attribute__((ext_vector_type(8)))  float  v8f;
typedef __attribute__((ext_vector_type(4)))  unsigned int u32x4;
typedef __attribute__((ext_vector_type(2)))  unsigned int u32x2;

union Frag {
    v16bf v;
    unsigned int u[8];
    u32x4 q[2];
    bf16 h[16];
};

__device__ __forceinline__ v8f wmma_bf16(v16bf a, v16bf b, v8f c) {
    return __builtin_amdgcn_wmma_f32_16x16x32_bf16(
        /*neg_a=*/false, a, /*neg_b=*/false, b,
        /*c_mod=*/(short)0, c, /*reuse_a=*/false, /*reuse_b=*/false);
}

// B fragment (32x16, K x N): lane&15 = column N; VGPR0-3: K=off..off+7,
// VGPR4-7: +16. base already includes +(lane>>4)*8.
__device__ __forceinline__ v16bf load_bfrag_g(const bf16* base) {
    Frag f;
    f.q[0] = *(const u32x4*)(base);
    f.q[1] = *(const u32x4*)(base + 16);
    return f.v;
}

// A fragment (16x32, M x K) from a row-major global matrix.
// base = row ptr + k0 + (half?8:0). VGPR0-3: K=off..off+7, VGPR4-7: +16.
__device__ __forceinline__ v16bf load_afrag_g(const bf16* base) {
    Frag f;
    f.q[0] = *(const u32x4*)(base);
    f.q[1] = *(const u32x4*)(base + 16);
    return f.v;
}

/* ------------------------ conversion kernels ------------------------ */

__global__ __launch_bounds__(256) void cvt_x_kernel(const float* __restrict__ x,
                                                    bf16* __restrict__ xb) {
    int idx = blockIdx.x * 256 + threadIdx.x;
    if (idx < NUM_TOK * DMODEL) xb[idx] = (bf16)x[idx];
}

// per-expert transpose + fp32->bf16:  in [z][R][C] -> out [z][C][R]
__global__ __launch_bounds__(256) void cvt_tr_kernel(const float* __restrict__ in,
                                                     bf16* __restrict__ out,
                                                     int R, int C) {
    __shared__ float t[32][33];
    const float* ein = in  + (size_t)blockIdx.z * R * C;
    bf16*       eout = out + (size_t)blockIdx.z * R * C;
    int bc = blockIdx.x * 32;
    int br = blockIdx.y * 32;
    int tid = threadIdx.x;
#pragma unroll
    for (int p = 0; p < 4; p++) {
        int r = p * 8 + (tid >> 5), c = tid & 31;
        t[r][c] = ein[(size_t)(br + r) * C + bc + c];
    }
    __syncthreads();
#pragma unroll
    for (int p = 0; p < 4; p++) {
        int r = p * 8 + (tid >> 5), c = tid & 31;
        eout[(size_t)(bc + r) * R + br + c] = (bf16)t[c][r];
    }
}

/* --------------------------- router path --------------------------- */

__global__ __launch_bounds__(256) void init_kernel(int* __restrict__ stok,
                                                   float* __restrict__ swt,
                                                   int* __restrict__ counts,
                                                   int* __restrict__ cursor) {
    int idx = blockIdx.x * 256 + threadIdx.x;
    if (idx < SLOTS_PAD) { stok[idx] = -1; swt[idx] = 0.f; }
    if (idx < NEXP)      { counts[idx] = 0; cursor[idx] = 0; }
}

__global__ __launch_bounds__(128) void router_kernel(const float* __restrict__ x,
                                                     const float* __restrict__ rw,
                                                     const float* __restrict__ rb,
                                                     int* __restrict__ topi,
                                                     float* __restrict__ topp,
                                                     int* __restrict__ counts) {
    __shared__ float red[128 * NEXP];
    __shared__ float lg[NEXP];
    int t = blockIdx.x, tid = threadIdx.x;
    const float* xr = x + (size_t)t * DMODEL;
    float la[NEXP];
#pragma unroll
    for (int e = 0; e < NEXP; e++) la[e] = 0.f;
    for (int d = tid; d < DMODEL; d += 128) {
        float xv = xr[d];
#pragma unroll
        for (int e = 0; e < NEXP; e++) la[e] += xv * rw[e * DMODEL + d];
    }
#pragma unroll
    for (int e = 0; e < NEXP; e++) red[tid * NEXP + e] = la[e];
    __syncthreads();
    if (tid < NEXP) {
        float s = 0.f;
        for (int j = 0; j < 128; j++) s += red[j * NEXP + tid];
        lg[tid] = s + rb[tid];
    }
    __syncthreads();
    if (tid == 0) {
        float p[NEXP];
        float mx = lg[0];
#pragma unroll
        for (int e = 1; e < NEXP; e++) mx = fmaxf(mx, lg[e]);
        float s = 0.f;
#pragma unroll
        for (int e = 0; e < NEXP; e++) { p[e] = __expf(lg[e] - mx); s += p[e]; }
        float invs = 1.f / s;
#pragma unroll
        for (int e = 0; e < NEXP; e++) p[e] *= invs;
        int i0 = 0;
#pragma unroll
        for (int e = 1; e < NEXP; e++) if (p[e] > p[i0]) i0 = e;
        int i1 = -1;
#pragma unroll
        for (int e = 0; e < NEXP; e++) {
            if (e == i0) continue;
            if (i1 < 0 || p[e] > p[i1]) i1 = e;
        }
        float inv2 = 1.f / (p[i0] + p[i1]);
        topi[2 * t] = i0;           topi[2 * t + 1] = i1;
        topp[2 * t] = p[i0] * inv2; topp[2 * t + 1] = p[i1] * inv2;
        atomicAdd(&counts[i0], 1);
        atomicAdd(&counts[i1], 1);
    }
}

__global__ void offsets_kernel(const int* __restrict__ counts, int* __restrict__ poff) {
    if (blockIdx.x == 0 && threadIdx.x == 0) {
        int o = 0;
        poff[0] = 0;
        for (int e = 0; e < NEXP; e++) {
            o += (counts[e] + 31) & ~31;   // 32-align each bucket (matches 32-row M tile)
            poff[e + 1] = o;
        }
    }
}

__global__ __launch_bounds__(256) void scatter_kernel(const int* __restrict__ topi,
                                                      const float* __restrict__ topp,
                                                      const int* __restrict__ poff,
                                                      int* __restrict__ cursor,
                                                      int* __restrict__ stok,
                                                      float* __restrict__ swt,
                                                      int* __restrict__ sof) {
    int t = blockIdx.x * 256 + threadIdx.x;
    if (t >= NUM_TOK) return;
#pragma unroll
    for (int k = 0; k < 2; k++) {
        int e = topi[2 * t + k];
        int idx = atomicAdd(&cursor[e], 1);
        int s = poff[e] + idx;
        stok[s] = t;
        swt[s] = topp[2 * t + k];
        sof[2 * t + k] = s;
    }
}

/* ---------------------- expert FFN (WMMA bf16) ---------------------- */

// grid (MTILES, DINNER/128), block 256 = 8 waves.
// Block tile: 32 slots x 128 F-cols. Wave w owns 16 cols; computes gate+up for
// two 16-row M subtiles sharing the same B fragments. LDS A is double-buffered
// and the staging GLOBAL load is software-pipelined one iteration ahead so its
// wait lands a full K-step (4 WMMAs) after issue.
__global__ __launch_bounds__(256) void ffn1_kernel(const bf16* __restrict__ xb,
                                                   const bf16* __restrict__ gwt,  // [E][F][D]
                                                   const bf16* __restrict__ uwt,  // [E][F][D]
                                                   const int* __restrict__ stok,
                                                   const int* __restrict__ poff,
                                                   bf16* __restrict__ hbuf) {     // [SLOTS_PAD][F]
    __shared__ bf16 As[2][32][40];

    int slot0 = blockIdx.x * 32;
    int tid = threadIdx.x;
    int lane = tid & 31, w = tid >> 5;
    int ncol = lane & 15, half = lane >> 4;
    int koffB = half * 8, koffA = half * 8;
    int nbase = blockIdx.y * 128 + w * 16;

    int e = 0;
#pragma unroll
    for (int i = 1; i < NEXP; i++) if (slot0 >= poff[i]) e = i;

    // staging assignment: thread loads 4 bf16 (8B) of row (tid>>3), cols (tid&7)*4..
    int srow = tid >> 3, scol = (tid & 7) * 4;
    int tk = stok[slot0 + srow];
    const bf16* srcRow = xb + (size_t)(tk < 0 ? 0 : tk) * DMODEL + scol;
    bool valid = (tk >= 0);
    const u32x2 zero2 = {0u, 0u};

    const bf16* gRow = gwt + ((size_t)e * DINNER + nbase + ncol) * DMODEL + koffB;
    const bf16* uRow = uwt + ((size_t)e * DINNER + nbase + ncol) * DMODEL + koffB;

    v8f cg0 = {}, cg1 = {}, cu0 = {}, cu1 = {};

    // prologue: stage chunk 0 into buffer 0, then pre-load chunk 1 into a register
    {
        u32x2 v0 = *(const u32x2*)(srcRow);
        *(u32x2*)&As[0][srow][scol] = valid ? v0 : zero2;
    }
    __syncthreads();
    u32x2 sv = *(const u32x2*)(srcRow + 32);   // chunk for k0=32 (dead value if DMODEL==32)

    for (int k0 = 0; k0 < DMODEL; k0 += 32) {
        int cur = (k0 >> 5) & 1, nxt = cur ^ 1;
        if (k0 + 32 < DMODEL) {     // store the pre-loaded next chunk
            *(u32x2*)&As[nxt][srow][scol] = valid ? sv : zero2;
            __builtin_prefetch(gRow + k0 + 32, 0, 1);
            __builtin_prefetch(uRow + k0 + 32, 0, 1);
        }
        if (k0 + 64 < DMODEL) {     // issue the load for the chunk after that
            sv = *(const u32x2*)(srcRow + k0 + 64);
        }

        Frag a0, a1;
#pragma unroll
        for (int r = 0; r < 4; r++) {
            a0.u[r]     = *(const unsigned int*)&As[cur][ncol][koffA + 2 * r];
            a0.u[4 + r] = *(const unsigned int*)&As[cur][ncol][16 + koffA + 2 * r];
            a1.u[r]     = *(const unsigned int*)&As[cur][16 + ncol][koffA + 2 * r];
            a1.u[4 + r] = *(const unsigned int*)&As[cur][16 + ncol][16 + koffA + 2 * r];
        }
        v16bf bg = load_bfrag_g(gRow + k0);
        v16bf bu = load_bfrag_g(uRow + k0);
        cg0 = wmma_bf16(a0.v, bg, cg0);
        cg1 = wmma_bf16(a1.v, bg, cg1);
        cu0 = wmma_bf16(a0.v, bu, cu0);
        cu1 = wmma_bf16(a1.v, bu, cu1);
        __syncthreads();
    }

    // h = silu(g) * u -> bf16. C layout: row M = r + half*8 (+16 for subtile 1).
#pragma unroll
    for (int r = 0; r < 8; r++) {
        int M = r + half * 8;
        float g0 = cg0[r], u0 = cu0[r];
        float g1 = cg1[r], u1 = cu1[r];
        float h0 = (g0 / (1.f + __expf(-g0))) * u0;
        float h1 = (g1 / (1.f + __expf(-g1))) * u1;
        hbuf[(size_t)(slot0 + M) * DINNER + nbase + ncol]      = (bf16)h0;
        hbuf[(size_t)(slot0 + 16 + M) * DINNER + nbase + ncol] = (bf16)h1;
    }
}

// grid (MTILES, DMODEL/256), block 256 = 8 waves.
// Block tile: 32 slots x 256 D-cols. Wave w owns 32 cols (2 N tiles) and
// 2 M subtiles -> 4 WMMA per K-step with shared A/B fragments.
__global__ __launch_bounds__(256) void ffn2_kernel(const bf16* __restrict__ hbuf, // [SLOTS_PAD][F]
                                                   const bf16* __restrict__ dwt,  // [E][D][F]
                                                   const float* __restrict__ swt,
                                                   const int* __restrict__ poff,
                                                   float* __restrict__ yc) {      // [SLOTS_PAD][D]
    __shared__ float wts[32];

    int slot0 = blockIdx.x * 32;
    int tid = threadIdx.x;
    int lane = tid & 31, w = tid >> 5;
    int ncol = lane & 15, half = lane >> 4;
    int koffB = half * 8, koffA = half * 8;
    int dbase = blockIdx.y * 256 + w * 32;

    int e = 0;
#pragma unroll
    for (int i = 1; i < NEXP; i++) if (slot0 >= poff[i]) e = i;

    if (tid < 32) wts[tid] = swt[slot0 + tid];
    __syncthreads();

    const bf16* aRow0 = hbuf + (size_t)(slot0 + ncol) * DINNER + koffA;
    const bf16* aRow1 = hbuf + (size_t)(slot0 + 16 + ncol) * DINNER + koffA;
    const bf16* bRow0 = dwt + ((size_t)e * DMODEL + dbase + ncol) * DINNER + koffB;
    const bf16* bRow1 = dwt + ((size_t)e * DMODEL + dbase + 16 + ncol) * DINNER + koffB;

    v8f c00 = {}, c01 = {}, c10 = {}, c11 = {};
    for (int k0 = 0; k0 < DINNER; k0 += 32) {
        v16bf a0 = load_afrag_g(aRow0 + k0);
        v16bf a1 = load_afrag_g(aRow1 + k0);
        v16bf b0 = load_bfrag_g(bRow0 + k0);
        v16bf b1 = load_bfrag_g(bRow1 + k0);
        __builtin_prefetch(bRow0 + k0 + 32, 0, 1);
        __builtin_prefetch(bRow1 + k0 + 32, 0, 1);
        c00 = wmma_bf16(a0, b0, c00);
        c01 = wmma_bf16(a0, b1, c01);
        c10 = wmma_bf16(a1, b0, c10);
        c11 = wmma_bf16(a1, b1, c11);
    }

#pragma unroll
    for (int r = 0; r < 8; r++) {
        int M = r + half * 8;
        float w0 = wts[M], w1 = wts[16 + M];
        yc[(size_t)(slot0 + M) * DMODEL + dbase + ncol]           = c00[r] * w0;
        yc[(size_t)(slot0 + M) * DMODEL + dbase + 16 + ncol]      = c01[r] * w0;
        yc[(size_t)(slot0 + 16 + M) * DMODEL + dbase + ncol]      = c10[r] * w1;
        yc[(size_t)(slot0 + 16 + M) * DMODEL + dbase + 16 + ncol] = c11[r] * w1;
    }
}

__global__ __launch_bounds__(256) void combine_kernel(const float* __restrict__ yc,
                                                      const int* __restrict__ sof,
                                                      float* __restrict__ out) {
    int idx = blockIdx.x * 256 + threadIdx.x;
    if (idx >= NUM_TOK * DMODEL) return;
    int t = idx >> 10;
    int d = idx & (DMODEL - 1);
    int s0 = sof[2 * t], s1 = sof[2 * t + 1];
    out[idx] = yc[(size_t)s0 * DMODEL + d] + yc[(size_t)s1 * DMODEL + d];
}

/* ----------------------------- launcher ----------------------------- */

extern "C" void kernel_launch(void* const* d_in, const int* in_sizes, int n_in,
                              void* d_out, int out_size, void* d_ws, size_t ws_size,
                              hipStream_t stream) {
    const float* x   = (const float*)d_in[0];
    const float* rw  = (const float*)d_in[1];
    const float* rb  = (const float*)d_in[2];
    const float* gw  = (const float*)d_in[3];
    const float* uw  = (const float*)d_in[4];
    const float* dw  = (const float*)d_in[5];
    float* out = (float*)d_out;

    char* base = (char*)d_ws;
    size_t off = 0;
    auto take = [&](size_t bytes) -> char* {
        char* p = base + off;
        off = (off + bytes + 255) & ~(size_t)255;
        return p;
    };
    bf16*  gwt   = (bf16*)take((size_t)NEXP * DINNER * DMODEL * 2);
    bf16*  uwt   = (bf16*)take((size_t)NEXP * DINNER * DMODEL * 2);
    bf16*  dwt   = (bf16*)take((size_t)NEXP * DMODEL * DINNER * 2);
    bf16*  xb    = (bf16*)take((size_t)NUM_TOK * DMODEL * 2);
    bf16*  hbuf  = (bf16*)take((size_t)SLOTS_PAD * DINNER * 2);
    float* yc    = (float*)take((size_t)SLOTS_PAD * DMODEL * 4);
    int*   topi  = (int*)take((size_t)NUM_TOK * 2 * 4);
    float* topp  = (float*)take((size_t)NUM_TOK * 2 * 4);
    int*   stok  = (int*)take((size_t)SLOTS_PAD * 4);
    float* swt   = (float*)take((size_t)SLOTS_PAD * 4);
    int*   sof   = (int*)take((size_t)NUM_TOK * 2 * 4);
    int*   counts= (int*)take(NEXP * 4);
    int*   cursor= (int*)take(NEXP * 4);
    int*   poff  = (int*)take((NEXP + 1) * 4);

    init_kernel<<<(SLOTS_PAD + 255) / 256, 256, 0, stream>>>(stok, swt, counts, cursor);

    cvt_x_kernel<<<(NUM_TOK * DMODEL) / 256, 256, 0, stream>>>(x, xb);
    cvt_tr_kernel<<<dim3(DINNER / 32, DMODEL / 32, NEXP), 256, 0, stream>>>(gw, gwt, DMODEL, DINNER);
    cvt_tr_kernel<<<dim3(DINNER / 32, DMODEL / 32, NEXP), 256, 0, stream>>>(uw, uwt, DMODEL, DINNER);
    cvt_tr_kernel<<<dim3(DMODEL / 32, DINNER / 32, NEXP), 256, 0, stream>>>(dw, dwt, DINNER, DMODEL);

    router_kernel<<<NUM_TOK, 128, 0, stream>>>(x, rw, rb, topi, topp, counts);
    offsets_kernel<<<1, 32, 0, stream>>>(counts, poff);
    scatter_kernel<<<(NUM_TOK + 255) / 256, 256, 0, stream>>>(topi, topp, poff, cursor,
                                                              stok, swt, sof);

    ffn1_kernel<<<dim3(MTILES, DINNER / 128), 256, 0, stream>>>(xb, gwt, uwt, stok, poff, hbuf);
    ffn2_kernel<<<dim3(MTILES, DMODEL / 256), 256, 0, stream>>>(hbuf, dwt, swt, poff, yc);

    combine_kernel<<<(NUM_TOK * DMODEL) / 256, 256, 0, stream>>>(yc, sof, out);
}
